// MyGNNLayer_26688926777933
// MI455X (gfx1250) — matrix operand order repaired and checked by reading, not compiled, not run on hardware
//
#include <hip/hip_runtime.h>
#include <hip/hip_bf16.h>
#include <math.h>

typedef __attribute__((ext_vector_type(16))) _Float16 v16h;
typedef __attribute__((ext_vector_type(8)))  float    v8f;
typedef __attribute__((ext_vector_type(4)))  float    f32x4;

#define EW 2   // waves per block, edge kernel (M=32 tiles)
#define NW 8   // waves per block, node kernel (M=16 tiles)
#define GW 8   // waves per block, global kernel

// ---------------------------------------------------------------------------
// exact (erf) GELU, matching jax.nn.gelu(approximate=False)
__device__ __forceinline__ float gelu(float v) {
    return 0.5f * v * (1.0f + erff(v * 0.70710678118654752f));
}

// ---------------------------------------------------------------------------
// A-fragment loader for V_WMMA_F32_16X16X32_F16.
// ISA 16-bit A 16x32 layout: lane (m = lane&15, sel = lane>>4) holds
// K = {off..off+7} in elems 0..7 and K = {off+16..off+23} in elems 8..15,
// where off = chunk_base + sel*8.  p points at row_base + off.
__device__ __forceinline__ v16h cvtA(const float* p) {
    f32x4 a0 = *(const f32x4*)(p);
    f32x4 a1 = *(const f32x4*)(p + 4);
    f32x4 a2 = *(const f32x4*)(p + 16);
    f32x4 a3 = *(const f32x4*)(p + 20);
    v16h r;
    r[0]=(_Float16)a0[0]; r[1]=(_Float16)a0[1]; r[2]=(_Float16)a0[2]; r[3]=(_Float16)a0[3];
    r[4]=(_Float16)a1[0]; r[5]=(_Float16)a1[1]; r[6]=(_Float16)a1[2]; r[7]=(_Float16)a1[3];
    r[8]=(_Float16)a2[0]; r[9]=(_Float16)a2[1]; r[10]=(_Float16)a2[2]; r[11]=(_Float16)a2[3];
    r[12]=(_Float16)a3[0]; r[13]=(_Float16)a3[1]; r[14]=(_Float16)a3[2]; r[15]=(_Float16)a3[3];
    return r;
}
__device__ __forceinline__ v16h cvtAs(const float* p, float s) {
    f32x4 a0 = *(const f32x4*)(p);
    f32x4 a1 = *(const f32x4*)(p + 4);
    f32x4 a2 = *(const f32x4*)(p + 16);
    f32x4 a3 = *(const f32x4*)(p + 20);
    v16h r;
    r[0]=(_Float16)(a0[0]*s); r[1]=(_Float16)(a0[1]*s); r[2]=(_Float16)(a0[2]*s); r[3]=(_Float16)(a0[3]*s);
    r[4]=(_Float16)(a1[0]*s); r[5]=(_Float16)(a1[1]*s); r[6]=(_Float16)(a1[2]*s); r[7]=(_Float16)(a1[3]*s);
    r[8]=(_Float16)(a2[0]*s); r[9]=(_Float16)(a2[1]*s); r[10]=(_Float16)(a2[2]*s); r[11]=(_Float16)(a2[3]*s);
    r[12]=(_Float16)(a3[0]*s); r[13]=(_Float16)(a3[1]*s); r[14]=(_Float16)(a3[2]*s); r[15]=(_Float16)(a3[3]*s);
    return r;
}

// ---------------------------------------------------------------------------
// Stage a [KC*32, 64] f32 row-major weight matrix into LDS in B-fragment
// layout.  B 32x16 16-bit layout (matching A): lane (n =
// lane&15, sel = lane>>4) elem t holds K = kc*32 + sel*8 + (t&7) + (t>>3)*16, col nt*16+n.
// Each wave then reads its fragment as one contiguous 32B LDS load.
__device__ __forceinline__ void stage_frags(const float* __restrict__ W,
                                            _Float16* dst, int KC,
                                            int tid, int nth) {
    const int total = KC * 2048;   // KC * 4 ntiles * 32 lanes * 16 elems
    for (int t = tid; t < total; t += nth) {
        int elem = t & 15;
        int lane = (t >> 4) & 31;
        int nt   = (t >> 9) & 3;
        int kc   = t >> 11;
        int n = nt * 16 + (lane & 15);
        int k = kc * 32 + ((lane >> 4) << 3) + (elem & 7) + ((elem >> 3) << 4);
        dst[t] = (_Float16)W[k * 64 + n];
    }
}

__device__ __forceinline__ v8f wmma_step(v16h a, v16h b, v8f c) {
    return __builtin_amdgcn_wmma_f32_16x16x32_f16(false, a, false, b,
                                                  (short)0, c, false, false);
}

// ---------------------------------------------------------------------------
__global__ void zero_kernel(float* __restrict__ p, long long n) {
    long long i = (long long)blockIdx.x * blockDim.x + threadIdx.x;
    long long stride = (long long)gridDim.x * blockDim.x;
    for (; i < n; i += stride) p[i] = 0.0f;
}

// ---------------------------------------------------------------------------
// Edge kernel: per 32-edge tile (two 16-row WMMA sub-tiles sharing B frags)
//   e = GELU([x[row] | x[col] | ea | u[batch[row]]] @ We + bE)
//   ea_new = LN(e + ea)                                  -> d_out segment
//   m = GELU([x[col] | ea_new] @ Wn1 + bN1)              -> atomic agg/cnt
__global__ __launch_bounds__(EW * 32) void edge_kernel(
    const float* __restrict__ x,  const int* __restrict__ ei,
    const float* __restrict__ ea, const float* __restrict__ u,
    const int* __restrict__ batch,
    const float* __restrict__ We,  const float* __restrict__ bE,
    const float* __restrict__ gE,  const float* __restrict__ bE2,
    const float* __restrict__ Wn1, const float* __restrict__ bN1,
    float* __restrict__ ea_out,
    float* __restrict__ agg, float* __restrict__ cnt,
    int E, int nTiles)
{
    __shared__ _Float16 sW[16384];                       // We frags (32KB), reused for Wn1
    __shared__ __align__(16) float sTile[EW][2048];      // 32x64 f32 per wave
    __shared__ int sRow[EW][32];

    const int tid  = threadIdx.x;
    const int lane = tid & 31;
    const int w    = tid >> 5;
    const int tile = blockIdx.x * EW + w;
    const int m    = lane & 15;
    const int sel  = lane >> 4;
    const bool active = (tile < nTiles);

    stage_frags(We, sW, 8, tid, EW * 32);
    __syncthreads();

    int e0 = 0, e1 = 0, row0 = 0, row1 = 0, col0 = 0, col1 = 0, g0 = 0, g1 = 0;
    if (active) {
        e0 = tile * 32 + m;         e1 = e0 + 16;
        row0 = ei[e0];              row1 = ei[e1];
        col0 = ei[E + e0];          col1 = ei[E + e1];
        g0 = batch[row0];           g1 = batch[row1];
        if (lane < 16) { sRow[w][lane] = row0; sRow[w][16 + lane] = row1; }
    }
    const float* sa0 = x  + (size_t)row0 * 64;
    const float* sb0 = x  + (size_t)col0 * 64;
    const float* sc0 = ea + (size_t)e0   * 64;
    const float* sd0 = u  + (size_t)g0   * 64;
    const float* sa1 = x  + (size_t)row1 * 64;
    const float* sb1 = x  + (size_t)col1 * 64;
    const float* sc1 = ea + (size_t)e1   * 64;
    const float* sd1 = u  + (size_t)g1   * 64;

    if (active) {
        v8f acc0[4] = {}, acc1[4] = {};
        const _Float16* bl = sW + lane * 16;
        v16h bnext[4];
        #pragma unroll
        for (int nt = 0; nt < 4; ++nt) bnext[nt] = *(const v16h*)(bl + nt * 512);
        #pragma unroll
        for (int kc = 0; kc < 8; ++kc) {
            v16h bcur[4];
            #pragma unroll
            for (int nt = 0; nt < 4; ++nt) bcur[nt] = bnext[nt];
            if (kc < 7) {             // prefetch next K-chunk's B frags
                #pragma unroll
                for (int nt = 0; nt < 4; ++nt)
                    bnext[nt] = *(const v16h*)(bl + (kc + 1) * 2048 + nt * 512);
            }
            const float* s0 = (kc < 2) ? sa0 : (kc < 4) ? sb0 : (kc < 6) ? sc0 : sd0;
            const float* s1 = (kc < 2) ? sa1 : (kc < 4) ? sb1 : (kc < 6) ? sc1 : sd1;
            int off = ((kc & 1) << 5) + (sel << 3);
            v16h a0 = cvtA(s0 + off);
            v16h a1 = cvtA(s1 + off);
            #pragma unroll
            for (int nt = 0; nt < 4; ++nt) {
                acc0[nt] = wmma_step(a0, bcur[nt], acc0[nt]);
                acc1[nt] = wmma_step(a1, bcur[nt], acc1[nt]);
            }
        }
        // epilogue: GELU(acc + bE) -> tile LDS (rows 0..15 sub0, 16..31 sub1)
        #pragma unroll
        for (int nt = 0; nt < 4; ++nt) {
            int n = nt * 16 + m;
            float bv = bE[n];
            #pragma unroll
            for (int v = 0; v < 8; ++v) {
                sTile[w][(v + sel * 8) * 64 + n]        = gelu(acc0[nt][v] + bv);
                sTile[w][(16 + v + sel * 8) * 64 + n]   = gelu(acc1[nt][v] + bv);
            }
        }
        // residual + LayerNorm: half-wave per row, shfl_xor(16) combine
        #pragma unroll
        for (int sub = 0; sub < 2; ++sub) {
            int r = sub * 16 + m;
            size_t erow = (size_t)(sub ? e1 : e0) * 64;
            int c0 = sel * 32;
            float vals[32];
            float s1 = 0.f, s2 = 0.f;
            #pragma unroll
            for (int c = 0; c < 32; ++c) {
                float vv = sTile[w][r * 64 + c0 + c] + ea[erow + c0 + c];
                vals[c] = vv; s1 += vv; s2 += vv * vv;
            }
            s1 += __shfl_xor(s1, 16);
            s2 += __shfl_xor(s2, 16);
            float mean = s1 * (1.0f / 64.0f);
            float var  = s2 * (1.0f / 64.0f) - mean * mean;
            float rstd = rsqrtf(var + 1e-5f);
            #pragma unroll
            for (int c = 0; c < 32; ++c) {
                int cc = c0 + c;
                float o = (vals[c] - mean) * rstd * gE[cc] + bE2[cc];
                ea_out[erow + cc] = o;
                sTile[w][r * 64 + cc] = o;     // feeds node-MLP1
            }
        }
    }
    __syncthreads();
    stage_frags(Wn1, sW, 4, tid, EW * 32);     // overwrite with Wn1 frags
    __syncthreads();
    if (active) {
        v8f acc0[4] = {}, acc1[4] = {};
        const float* t0 = &sTile[w][m * 64];
        const float* t1 = &sTile[w][(16 + m) * 64];
        const _Float16* bl = sW + lane * 16;
        v16h bnext[4];
        #pragma unroll
        for (int nt = 0; nt < 4; ++nt) bnext[nt] = *(const v16h*)(bl + nt * 512);
        #pragma unroll
        for (int kc = 0; kc < 4; ++kc) {
            v16h bcur[4];
            #pragma unroll
            for (int nt = 0; nt < 4; ++nt) bcur[nt] = bnext[nt];
            if (kc < 3) {
                #pragma unroll
                for (int nt = 0; nt < 4; ++nt)
                    bnext[nt] = *(const v16h*)(bl + (kc + 1) * 2048 + nt * 512);
            }
            int off = ((kc & 1) << 5) + (sel << 3);
            v16h a0 = (kc < 2) ? cvtA(sb0 + off) : cvtA(t0 + off);
            v16h a1 = (kc < 2) ? cvtA(sb1 + off) : cvtA(t1 + off);
            #pragma unroll
            for (int nt = 0; nt < 4; ++nt) {
                acc0[nt] = wmma_step(a0, bcur[nt], acc0[nt]);
                acc1[nt] = wmma_step(a1, bcur[nt], acc1[nt]);
            }
        }
        #pragma unroll
        for (int nt = 0; nt < 4; ++nt) {
            int n = nt * 16 + m;
            float bv = bN1[n];
            #pragma unroll
            for (int v = 0; v < 8; ++v) {
                int rM0 = sRow[w][v + sel * 8];
                int rM1 = sRow[w][16 + v + sel * 8];
                atomicAdd(agg + (size_t)rM0 * 64 + n, gelu(acc0[nt][v] + bv));
                atomicAdd(agg + (size_t)rM1 * 64 + n, gelu(acc1[nt][v] + bv));
            }
        }
        if (lane < 16) {
            atomicAdd(cnt + row0, 1.0f);
            atomicAdd(cnt + row1, 1.0f);
        }
    }
}

// ---------------------------------------------------------------------------
// Node kernel: x_new = LN(GELU([agg/cnt | u[batch]] @ Wn2 + bN2) + x)
// plus scatter into gsum/gcnt for the global model.
__global__ __launch_bounds__(NW * 32) void node_kernel(
    const float* __restrict__ x, const float* __restrict__ u,
    const int* __restrict__ batch,
    const float* __restrict__ agg, const float* __restrict__ cnt,
    const float* __restrict__ Wn2, const float* __restrict__ bN2,
    const float* __restrict__ gN,  const float* __restrict__ bN,
    float* __restrict__ x_out, float* __restrict__ gsum, float* __restrict__ gcnt,
    int N, int nTiles)
{
    __shared__ _Float16 sW[8192];
    __shared__ __align__(16) float sTile[NW][1024];

    const int tid  = threadIdx.x;
    const int lane = tid & 31;
    const int w    = tid >> 5;
    const int tile = blockIdx.x * NW + w;
    const int m    = lane & 15;
    const int sel  = lane >> 4;
    const bool active = (tile < nTiles);

    stage_frags(Wn2, sW, 4, tid, NW * 32);
    __syncthreads();
    if (!active) return;

    int node = tile * 16 + m;
    int g = batch[node];
    float inv = 1.0f / fmaxf(cnt[node], 1.0f);
    const float* segA = agg + (size_t)node * 64;
    const float* segU = u   + (size_t)g    * 64;

    v8f acc[4] = {};
    const _Float16* bl = sW + lane * 16;
    v16h bnext[4];
    #pragma unroll
    for (int nt = 0; nt < 4; ++nt) bnext[nt] = *(const v16h*)(bl + nt * 512);
    #pragma unroll
    for (int kc = 0; kc < 4; ++kc) {
        v16h bcur[4];
        #pragma unroll
        for (int nt = 0; nt < 4; ++nt) bcur[nt] = bnext[nt];
        if (kc < 3) {
            #pragma unroll
            for (int nt = 0; nt < 4; ++nt)
                bnext[nt] = *(const v16h*)(bl + (kc + 1) * 2048 + nt * 512);
        }
        int off = ((kc & 1) << 5) + (sel << 3);
        v16h a = (kc < 2) ? cvtAs(segA + off, inv) : cvtA(segU + off);
        #pragma unroll
        for (int nt = 0; nt < 4; ++nt)
            acc[nt] = wmma_step(a, bcur[nt], acc[nt]);
    }
    #pragma unroll
    for (int nt = 0; nt < 4; ++nt) {
        int n = nt * 16 + m;
        float bv = bN2[n];
        #pragma unroll
        for (int v = 0; v < 8; ++v)
            sTile[w][(v + sel * 8) * 64 + n] = gelu(acc[nt][v] + bv);
    }
    {   // residual + LN; row m belongs to this lane's own node
        size_t nrow = (size_t)node * 64;
        int c0 = sel * 32;
        float vals[32];
        float s1 = 0.f, s2 = 0.f;
        #pragma unroll
        for (int c = 0; c < 32; ++c) {
            float vv = sTile[w][m * 64 + c0 + c] + x[nrow + c0 + c];
            vals[c] = vv; s1 += vv; s2 += vv * vv;
        }
        s1 += __shfl_xor(s1, 16);
        s2 += __shfl_xor(s2, 16);
        float mean = s1 * (1.0f / 64.0f);
        float var  = s2 * (1.0f / 64.0f) - mean * mean;
        float rstd = rsqrtf(var + 1e-5f);
        #pragma unroll
        for (int c = 0; c < 32; ++c) {
            int cc = c0 + c;
            float o = (vals[c] - mean) * rstd * gN[cc] + bN[cc];
            x_out[nrow + cc] = o;
            atomicAdd(gsum + (size_t)g * 64 + cc, o);
        }
        if (lane < 16) atomicAdd(gcnt + g, 1.0f);
    }
}

// ---------------------------------------------------------------------------
// Global kernel: u_new = LN(GELU([u | gsum/gcnt] @ Wg + bG) + u)
__global__ __launch_bounds__(GW * 32) void global_kernel(
    const float* __restrict__ u, const float* __restrict__ gsum,
    const float* __restrict__ gcnt,
    const float* __restrict__ Wg, const float* __restrict__ bG,
    const float* __restrict__ gG, const float* __restrict__ bG2,
    float* __restrict__ u_out, int G, int nTiles)
{
    __shared__ _Float16 sW[8192];
    __shared__ __align__(16) float sTile[GW][1024];

    const int tid  = threadIdx.x;
    const int lane = tid & 31;
    const int w    = tid >> 5;
    const int tile = blockIdx.x * GW + w;
    const int m    = lane & 15;
    const int sel  = lane >> 4;
    const bool active = (tile < nTiles);

    stage_frags(Wg, sW, 4, tid, GW * 32);
    __syncthreads();
    if (!active) return;

    int gr = tile * 16 + m;
    if (gr >= G) gr = G - 1;
    float inv = 1.0f / fmaxf(gcnt[gr], 1.0f);
    const float* segU = u    + (size_t)gr * 64;
    const float* segS = gsum + (size_t)gr * 64;

    v8f acc[4] = {};
    const _Float16* bl = sW + lane * 16;
    v16h bnext[4];
    #pragma unroll
    for (int nt = 0; nt < 4; ++nt) bnext[nt] = *(const v16h*)(bl + nt * 512);
    #pragma unroll
    for (int kc = 0; kc < 4; ++kc) {
        v16h bcur[4];
        #pragma unroll
        for (int nt = 0; nt < 4; ++nt) bcur[nt] = bnext[nt];
        if (kc < 3) {
            #pragma unroll
            for (int nt = 0; nt < 4; ++nt)
                bnext[nt] = *(const v16h*)(bl + (kc + 1) * 2048 + nt * 512);
        }
        int off = ((kc & 1) << 5) + (sel << 3);
        v16h a = (kc < 2) ? cvtA(segU + off) : cvtAs(segS + off, inv);
        #pragma unroll
        for (int nt = 0; nt < 4; ++nt)
            acc[nt] = wmma_step(a, bcur[nt], acc[nt]);
    }
    #pragma unroll
    for (int nt = 0; nt < 4; ++nt) {
        int n = nt * 16 + m;
        float bv = bG[n];
        #pragma unroll
        for (int v = 0; v < 8; ++v)
            sTile[w][(v + sel * 8) * 64 + n] = gelu(acc[nt][v] + bv);
    }
    {
        size_t grow = (size_t)gr * 64;
        int c0 = sel * 32;
        float vals[32];
        float s1 = 0.f, s2 = 0.f;
        #pragma unroll
        for (int c = 0; c < 32; ++c) {
            float vv = sTile[w][m * 64 + c0 + c] + u[grow + c0 + c];
            vals[c] = vv; s1 += vv; s2 += vv * vv;
        }
        s1 += __shfl_xor(s1, 16);
        s2 += __shfl_xor(s2, 16);
        float mean = s1 * (1.0f / 64.0f);
        float var  = s2 * (1.0f / 64.0f) - mean * mean;
        float rstd = rsqrtf(var + 1e-5f);
        if (tile * 16 + m < G) {
            #pragma unroll
            for (int c = 0; c < 32; ++c) {
                int cc = c0 + c;
                u_out[grow + cc] = (vals[c] - mean) * rstd * gG[cc] + bG2[cc];
            }
        }
    }
}

// ---------------------------------------------------------------------------
extern "C" void kernel_launch(void* const* d_in, const int* in_sizes, int n_in,
                              void* d_out, int out_size, void* d_ws, size_t ws_size,
                              hipStream_t stream) {
    const float* x     = (const float*)d_in[0];
    const int*   ei    = (const int*)  d_in[1];
    const float* ea    = (const float*)d_in[2];
    const float* u     = (const float*)d_in[3];
    const int*   batch = (const int*)  d_in[4];
    const float* We  = (const float*)d_in[5];
    const float* bE  = (const float*)d_in[6];
    const float* gE  = (const float*)d_in[7];
    const float* bE2 = (const float*)d_in[8];
    const float* Wn1 = (const float*)d_in[9];
    const float* bN1 = (const float*)d_in[10];
    const float* Wn2 = (const float*)d_in[11];
    const float* bN2 = (const float*)d_in[12];
    const float* gN  = (const float*)d_in[13];
    const float* bN  = (const float*)d_in[14];
    const float* Wg  = (const float*)d_in[15];
    const float* bG  = (const float*)d_in[16];
    const float* gG  = (const float*)d_in[17];
    const float* bG2 = (const float*)d_in[18];

    const int N = in_sizes[0] / 64;
    const int E = in_sizes[1] / 2;
    const int G = in_sizes[3] / 64;

    float* ws   = (float*)d_ws;
    float* agg  = ws;                               // [N,64]
    float* cnt  = agg  + (size_t)N * 64;            // [N]
    float* gsum = cnt  + N;                         // [G,64]
    float* gcnt = gsum + (size_t)G * 64;            // [G]
    long long zTotal = (long long)N * 64 + N + (long long)G * 64 + G;

    float* x_out  = (float*)d_out;                  // [N,64]
    float* ea_out = x_out  + (size_t)N * 64;        // [E,64]
    float* u_out  = ea_out + (size_t)E * 64;        // [G,64]

    zero_kernel<<<2048, 256, 0, stream>>>(ws, zTotal);

    int eTiles = (E + 31) / 32;
    edge_kernel<<<(eTiles + EW - 1) / EW, EW * 32, 0, stream>>>(
        x, ei, ea, u, batch, We, bE, gE, bE2, Wn1, bN1,
        ea_out, agg, cnt, E, eTiles);

    int nTilesN = (N + 15) / 16;
    node_kernel<<<(nTilesN + NW - 1) / NW, NW * 32, 0, stream>>>(
        x, u, batch, agg, cnt, Wn2, bN2, gN, bN,
        x_out, gsum, gcnt, N, nTilesN);

    int gTiles = (G + 15) / 16;
    global_kernel<<<(gTiles + GW - 1) / GW, GW * 32, 0, stream>>>(
        u, gsum, gcnt, Wg, bG, gG, bG2, u_out, G, gTiles);
}
